// TFEConv_38328288150260
// MI455X (gfx1250) — compile-verified, matrix operand order checked by reading, not run on hardware
//
#include <hip/hip_runtime.h>
#include <math.h>

#define NN 16384
#define KSPLIT 4
#define KCHUNK (NN / KSPLIT)
#define ROWS_PER_BLOCK 128

typedef __attribute__((ext_vector_type(16))) __bf16 v16bf;
typedef __attribute__((ext_vector_type(8))) float v8f;

union Frag {
    v16bf v;
    int4 q[2];
};

__device__ __forceinline__ unsigned short f2bf(float f) {
    unsigned u = __float_as_uint(f);
    unsigned r = u + 0x7FFFu + ((u >> 16) & 1u);
    return (unsigned short)(r >> 16);
}

__device__ __forceinline__ int pack2(float a, float b) {
    return (int)((unsigned)f2bf(a) | ((unsigned)f2bf(b) << 16));
}

// ---------------- one-time L_hat f32 -> bf16 ----------------
__global__ __launch_bounds__(256) void convert_L_bf16(const float* __restrict__ src,
                                                      unsigned short* __restrict__ dst) {
    size_t i = ((size_t)blockIdx.x * 256 + threadIdx.x) * 8;
    float4 a = *(const float4*)(src + i);
    float4 b = *(const float4*)(src + i + 4);
    int4 o = make_int4(pack2(a.x, a.y), pack2(a.z, a.w), pack2(b.x, b.y), pack2(b.z, b.w));
    *(int4*)(dst + i) = o;
}

// ---------------- build bf16 transposed panel Bt[c][r] = T[r][c] ----------------
__global__ __launch_bounds__(256) void prep_bt(const float* __restrict__ T,
                                               unsigned short* __restrict__ Bt) {
    int idx = blockIdx.x * 256 + threadIdx.x;  // 0 .. NN*32-1
    int r = idx >> 5, c = idx & 31;
    Bt[(size_t)c * NN + r] = f2bf(T[idx]);
}

// ---------------- LP = a0*X, HPc = b0*X ----------------
__global__ __launch_bounds__(256) void init_lp(const float* __restrict__ X, float* __restrict__ LP,
                                               float* __restrict__ HPc, float clp, float chp) {
    int i = blockIdx.x * 256 + threadIdx.x;
    float x = X[i];
    LP[i] = clp * x;
    HPc[i] = chp * x;
}

// ---- Tnext = sum(partials) + beta*Tprev ; LP += clp*Tnext ; HPc += chp*Tnext (deterministic) ----
__global__ __launch_bounds__(256) void reduce_next(const float* __restrict__ partial,
                                                   const float* __restrict__ Tprev, float beta,
                                                   float clp, float chp, float* __restrict__ Tnext,
                                                   float* __restrict__ LP, float* __restrict__ HPc) {
    int i = blockIdx.x * 256 + threadIdx.x;
    const size_t S = (size_t)NN * 32;
    float s = partial[i] + partial[S + i] + partial[2 * S + i] + partial[3 * S + i];
    s += beta * Tprev[i];
    Tnext[i] = s;
    LP[i] += clp * s;
    HPc[i] += chp * s;
}

// ---------------- main WMMA GEMM: partial = alpha * (A @ T) over a K-chunk ----------------
// A: NN x NN (bf16 row-major, or f32 row-major converted on the fly)
// Bt: bf16 transposed panel [32][NN]  (column n's K-values contiguous)
// Each wave: 32 rows x 32 cols, 4 C-tiles, 4x v_wmma_f32_16x16x32_bf16 per K=32 step.
template <bool ABF16>
__global__ __launch_bounds__(128) void cheb_gemm(const void* __restrict__ Aop,
                                                 const unsigned short* __restrict__ Bt,
                                                 float* __restrict__ partial, float alpha) {
    const int lane = threadIdx.x & 31;
    const int wave = threadIdx.x >> 5;
    const int laneLow = lane & 15;
    const int laneHalf = lane >> 4;
    const int rowBase = blockIdx.x * ROWS_PER_BLOCK + wave * 32;
    const int k0 = blockIdx.y * KCHUNK;

    v8f acc00 = {0, 0, 0, 0, 0, 0, 0, 0};
    v8f acc01 = acc00, acc10 = acc00, acc11 = acc00;

    // Per-lane K offset matching the ISA 16-bit A/B 16x32 fragment layout:
    // lanes 0-15 hold K 0..7 and 16..23; lanes 16-31 hold K 8..15 and 24..31.
    const size_t kLane = (size_t)k0 + (size_t)laneHalf * 8;
    const unsigned short* bp0 = Bt + (size_t)laneLow * NN + kLane;
    const unsigned short* bp1 = Bt + (size_t)(laneLow + 16) * NN + kLane;

    const unsigned short* ap0b = nullptr;
    const unsigned short* ap1b = nullptr;
    const float* ap0f = nullptr;
    const float* ap1f = nullptr;
    if (ABF16) {
        const unsigned short* A = (const unsigned short*)Aop;
        ap0b = A + (size_t)(rowBase + laneLow) * NN + kLane;
        ap1b = A + (size_t)(rowBase + 16 + laneLow) * NN + kLane;
    } else {
        const float* A = (const float*)Aop;
        ap0f = A + (size_t)(rowBase + laneLow) * NN + kLane;
        ap1f = A + (size_t)(rowBase + 16 + laneLow) * NN + kLane;
    }

#pragma unroll 2
    for (int kk = 0; kk < KCHUNK; kk += 32) {
        Frag a0, a1, b0, b1;
        b0.q[0] = *(const int4*)(bp0);
        b0.q[1] = *(const int4*)(bp0 + 16);
        b1.q[0] = *(const int4*)(bp1);
        b1.q[1] = *(const int4*)(bp1 + 16);
        if (ABF16) {
            a0.q[0] = *(const int4*)(ap0b);
            a0.q[1] = *(const int4*)(ap0b + 16);
            a1.q[0] = *(const int4*)(ap1b);
            a1.q[1] = *(const int4*)(ap1b + 16);
            ap0b += 32;
            ap1b += 32;
        } else {
            float4 x0 = *(const float4*)(ap0f), x1 = *(const float4*)(ap0f + 4);
            float4 x2 = *(const float4*)(ap0f + 16), x3 = *(const float4*)(ap0f + 20);
            a0.q[0] = make_int4(pack2(x0.x, x0.y), pack2(x0.z, x0.w), pack2(x1.x, x1.y),
                                pack2(x1.z, x1.w));
            a0.q[1] = make_int4(pack2(x2.x, x2.y), pack2(x2.z, x2.w), pack2(x3.x, x3.y),
                                pack2(x3.z, x3.w));
            float4 y0 = *(const float4*)(ap1f), y1 = *(const float4*)(ap1f + 4);
            float4 y2 = *(const float4*)(ap1f + 16), y3 = *(const float4*)(ap1f + 20);
            a1.q[0] = make_int4(pack2(y0.x, y0.y), pack2(y0.z, y0.w), pack2(y1.x, y1.y),
                                pack2(y1.z, y1.w));
            a1.q[1] = make_int4(pack2(y2.x, y2.y), pack2(y2.z, y2.w), pack2(y3.x, y3.y),
                                pack2(y3.z, y3.w));
            ap0f += 32;
            ap1f += 32;
        }
        bp0 += 32;
        bp1 += 32;
        acc00 = __builtin_amdgcn_wmma_f32_16x16x32_bf16(false, a0.v, false, b0.v, (short)0, acc00,
                                                        false, false);
        acc01 = __builtin_amdgcn_wmma_f32_16x16x32_bf16(false, a0.v, false, b1.v, (short)0, acc01,
                                                        false, false);
        acc10 = __builtin_amdgcn_wmma_f32_16x16x32_bf16(false, a1.v, false, b0.v, (short)0, acc10,
                                                        false, false);
        acc11 = __builtin_amdgcn_wmma_f32_16x16x32_bf16(false, a1.v, false, b1.v, (short)0, acc11,
                                                        false, false);
    }

    // C/D layout: VGPR j -> row (base + laneHalf*8 + j), col = laneLow (+16 for second N tile)
    float* out = partial + (size_t)blockIdx.y * ((size_t)NN * 32);
#pragma unroll
    for (int j = 0; j < 8; ++j) {
        int r0 = rowBase + laneHalf * 8 + j;
        int r1 = r0 + 16;
        out[(size_t)r0 * 32 + laneLow] = alpha * acc00[j];
        out[(size_t)r0 * 32 + 16 + laneLow] = alpha * acc01[j];
        out[(size_t)r1 * 32 + laneLow] = alpha * acc10[j];
        out[(size_t)r1 * 32 + 16 + laneLow] = alpha * acc11[j];
    }
}

// ---------------- fused epilogue: Z = [relu(LP Wlp+b), relu((X-HPc) Whp+b), X] Wf + bf ----------
__global__ __launch_bounds__(256) void epilogue_kernel(
    const float* __restrict__ LP, const float* __restrict__ HPc, const float* __restrict__ X,
    const float* __restrict__ Wlp, const float* __restrict__ blp, const float* __restrict__ Whp,
    const float* __restrict__ bhp, const float* __restrict__ Wf, const float* __restrict__ bfv,
    float* __restrict__ Z) {
    __shared__ float sWlp[32 * 32];
    __shared__ float sWhp[32 * 32];
    __shared__ float sWf[96 * 32];
    __shared__ float sblp[32], sbhp[32], sbf[32];
    int tid = threadIdx.x;
    for (int i = tid; i < 1024; i += 256) {
        sWlp[i] = Wlp[i];
        sWhp[i] = Whp[i];
    }
    for (int i = tid; i < 3072; i += 256) sWf[i] = Wf[i];
    if (tid < 32) {
        sblp[tid] = blp[tid];
        sbhp[tid] = bhp[tid];
        sbf[tid] = bfv[tid];
    }
    __syncthreads();

    int r = blockIdx.x * 256 + tid;
    if (r >= NN) return;
    float lp[32], hp[32], xr[32], z[32];
#pragma unroll
    for (int j = 0; j < 32; ++j) {
        lp[j] = LP[r * 32 + j];
        xr[j] = X[r * 32 + j];
        hp[j] = xr[j] - HPc[r * 32 + j];
    }
#pragma unroll
    for (int c = 0; c < 32; ++c) z[c] = sbf[c];
    for (int j = 0; j < 32; ++j) {
        float h = sblp[j];
#pragma unroll
        for (int i = 0; i < 32; ++i) h = fmaf(lp[i], sWlp[i * 32 + j], h);
        h = fmaxf(h, 0.f);
#pragma unroll
        for (int c = 0; c < 32; ++c) z[c] = fmaf(h, sWf[j * 32 + c], z[c]);
    }
    for (int j = 0; j < 32; ++j) {
        float h = sbhp[j];
#pragma unroll
        for (int i = 0; i < 32; ++i) h = fmaf(hp[i], sWhp[i * 32 + j], h);
        h = fmaxf(h, 0.f);
#pragma unroll
        for (int c = 0; c < 32; ++c) z[c] = fmaf(h, sWf[(32 + j) * 32 + c], z[c]);
    }
    for (int j = 0; j < 32; ++j) {
#pragma unroll
        for (int c = 0; c < 32; ++c) z[c] = fmaf(xr[j], sWf[(64 + j) * 32 + c], z[c]);
    }
#pragma unroll
    for (int c = 0; c < 32; ++c) Z[(size_t)r * 32 + c] = z[c];
}

// ---------------- host side ----------------
static double factd(int n) {
    double f = 1.0;
    for (int i = 2; i <= n; ++i) f *= (double)i;
    return f;
}
static double besselI(int k, double x) {
    double s = 0.0;
    for (int m = 0; m < 40; ++m) s += pow(x * 0.5, (double)(2 * m + k)) / (factd(m) * factd(m + k));
    return s;
}

extern "C" void kernel_launch(void* const* d_in, const int* in_sizes, int n_in, void* d_out,
                              int out_size, void* d_ws, size_t ws_size, hipStream_t stream) {
    const float* X = (const float*)d_in[0];
    const float* L = (const float*)d_in[1];
    const float* Wlp = (const float*)d_in[2];
    const float* blp = (const float*)d_in[3];
    const float* Whp = (const float*)d_in[4];
    const float* bhp = (const float*)d_in[5];
    const float* Wf = (const float*)d_in[6];
    const float* bfv = (const float*)d_in[7];
    float* Z = (float*)d_out;

    // Chebyshev heat-kernel coefficients (deterministic, input-independent).
    float alp[9], ahp[9];
    for (int k = 0; k <= 8; ++k) {
        double s = (k > 0 ? 2.0 : 1.0) * ((k & 1) ? -1.0 : 1.0);
        alp[k] = (float)(s * exp(-0.5) * besselI(k, 0.5));
        ahp[k] = (float)(s * exp(-1.5) * besselI(k, 1.5));
    }

    // Workspace carving (256B aligned slots).
    char* w = (char*)d_ws;
    size_t used = 0;
    auto take = [&](size_t bytes) -> void* {
        void* p = w + used;
        used += (bytes + 255) & ~(size_t)255;
        return p;
    };
    unsigned short* Bt = (unsigned short*)take((size_t)NN * 32 * 2);
    float* tbuf[3];
    for (int i = 0; i < 3; ++i) tbuf[i] = (float*)take((size_t)NN * 32 * 4);
    float* partial = (float*)take((size_t)KSPLIT * NN * 32 * 4);
    float* LPb = (float*)take((size_t)NN * 32 * 4);
    float* HPb = (float*)take((size_t)NN * 32 * 4);
    size_t smallUsed = used;
    bool useBf = (ws_size >= smallUsed + (size_t)NN * NN * 2 + 512);
    unsigned short* Lb = useBf ? (unsigned short*)take((size_t)NN * NN * 2) : nullptr;

    const int ETH = 256;
    dim3 gPanel(NN * 32 / ETH);  // 2048 blocks over the N x 32 panel

    if (useBf) {
        // Convert the loop-invariant 1 GiB L_hat once -> halves HBM traffic of all 8 GEMMs.
        convert_L_bf16<<<dim3((unsigned)((size_t)NN * NN / 8 / ETH)), ETH, 0, stream>>>(L, Lb);
    }
    init_lp<<<gPanel, ETH, 0, stream>>>(X, LPb, HPb, alp[0], ahp[0]);
    prep_bt<<<gPanel, ETH, 0, stream>>>(X, Bt);  // panel of T0 = X

    const float* pKm2 = X;  // T_{k-2}
    const float* pKm1 = X;  // T_{k-1}
    for (int k = 1; k <= 8; ++k) {
        float alpha = (k == 1) ? 1.f : 2.f;
        float beta = (k == 1) ? 0.f : -1.f;
        dim3 gg(NN / ROWS_PER_BLOCK, KSPLIT);  // 128 x 4 blocks, 2048 waves
        if (useBf)
            cheb_gemm<true><<<gg, 128, 0, stream>>>(Lb, Bt, partial, alpha);
        else
            cheb_gemm<false><<<gg, 128, 0, stream>>>(L, Bt, partial, alpha);
        float* Tn = tbuf[(k - 1) % 3];
        reduce_next<<<gPanel, ETH, 0, stream>>>(partial, pKm2, beta, alp[k], ahp[k], Tn, LPb, HPb);
        if (k < 8) prep_bt<<<gPanel, ETH, 0, stream>>>(Tn, Bt);
        pKm2 = pKm1;
        pKm1 = Tn;
    }

    epilogue_kernel<<<dim3(NN / 256), 256, 0, stream>>>(LPb, HPb, X, Wlp, blp, Whp, bhp, Wf, bfv,
                                                        Z);
}
